// ChannelWiseSpatialAttentLearning_54614804136626
// MI455X (gfx1250) — matrix operand order repaired and checked by reading, not compiled, not run on hardware
//
#include <hip/hip_runtime.h>
#include <hip/hip_bf16.h>
#include <math.h>

// ---------------------------------------------------------------------------
// CDNA5 (gfx1250) implementation.
// Stage 1 (dominant): conv3x3+bias+relu+global-avg-pool via bf16 WMMA implicit
//   GEMM. Loop nest keeps A (weight) fragments in registers across all 4 pixel
//   tiles (4 accumulator pairs) -> 4x less L2 weight traffic than A-per-tile.
// Stage 2 (tail): per-batch matvec chain + scalar CRF, fp32 VALU.
// ---------------------------------------------------------------------------

typedef __attribute__((ext_vector_type(16))) __bf16 v16bf;
typedef __attribute__((ext_vector_type(8)))  float  v8f;

static constexpr int   Bn   = 16;
static constexpr int   Cn   = 256;
static constexpr int   Hn   = 56;
static constexpr int   Wn   = 56;
static constexpr int   HP   = 58;           // padded (halo) spatial extent
static constexpr int   CIP  = 264;          // ci padded in LDS (bank-conflict break)
static constexpr size_t WB_BYTES = (size_t)9 * 256 * 256 * 2;            // 1,179,648
static constexpr size_t XB_BYTES = (size_t)Bn * HP * HP * Cn * 2;        // 27,557,888
static constexpr int   XS_BYTES  = HP * 3 * CIP * 2;                     // 91,872 LDS strip
static constexpr int   SMEM_B    = XS_BYTES + Cn * 4;                    // + f1part

__device__ __forceinline__ unsigned short f2bf(float f) {
    union { float f; unsigned u; } x; x.f = f;
    unsigned r = x.u + 0x7FFFu + ((x.u >> 16) & 1u);   // round-to-nearest-even
    return (unsigned short)(r >> 16);
}

// ---------------- prep: zero f1 partial sums (small region) ----------------
__global__ void k_zero(uint4* __restrict__ p, size_t n4) {
    size_t i = (size_t)blockIdx.x * blockDim.x + threadIdx.x;
    size_t stride = (size_t)gridDim.x * blockDim.x;
    uint4 z; z.x = z.y = z.z = z.w = 0u;
    for (; i < n4; i += stride) p[i] = z;
}

// ---------------- prep: zero ONLY the halo ring of xb ----------------------
// 228 segments per batch: row 0 (58), row 57 (58), cols 0/57 for rows 1..56.
__global__ void k_zero_halo(unsigned short* __restrict__ xb) {
    const int sg = blockIdx.x;
    const int b = sg / 228;
    const int r = sg % 228;
    int hh, ww;
    if (r < 58)       { hh = 0;  ww = r; }
    else if (r < 116) { hh = 57; ww = r - 58; }
    else { const int rr = r - 116; hh = 1 + (rr >> 1); ww = (rr & 1) ? 57 : 0; }
    uint4* p = (uint4*)(xb + (((size_t)b * HP + hh) * HP + ww) * Cn);
    uint4 z; z.x = z.y = z.z = z.w = 0u;
    if (threadIdx.x < 32) p[threadIdx.x] = z;   // 32 * 16B = 256 bf16
}

// ---------------- prep: x NCHW fp32 -> NHWC bf16 with halo -----------------
__global__ void k_xprep(const float* __restrict__ x, unsigned short* __restrict__ xb) {
    __shared__ float tile[64][57];
    const int blk = blockIdx.x;
    const int cb  = blk & 3;
    const int h   = (blk >> 2) % Hn;
    const int b   = blk / (4 * Hn);
    const int ci0 = cb * 64;
    for (int t = threadIdx.x; t < 64 * Wn; t += 256) {
        const int i = t / Wn, w = t - i * Wn;
        tile[i][w] = x[(((size_t)b * Cn + ci0 + i) * Hn + h) * Wn + w];
    }
    __syncthreads();
    for (int t = threadIdx.x; t < Wn * 64; t += 256) {
        const int w = t >> 6, i = t & 63;
        xb[(((size_t)b * HP + h + 1) * HP + (w + 1)) * Cn + ci0 + i] = f2bf(tile[i][w]);
    }
}

// ---------------- prep: w0_0 OIHW fp32 -> [tap][co][ci] bf16 ---------------
__global__ void k_wprep(const float* __restrict__ w0, unsigned short* __restrict__ wb) {
    const int gid = blockIdx.x * 256 + threadIdx.x;      // < 589824
    const int ci = gid & 255;
    const int co = (gid >> 8) & 255;
    const int s  = gid >> 16;                            // kh*3+kw
    wb[gid] = f2bf(w0[((size_t)co * Cn + ci) * 9 + s]);
}

// ---------------- main: conv + relu + spatial sum (WMMA) -------------------
__global__ __launch_bounds__(256)
void k_conv_pool(const unsigned short* __restrict__ xb,
                 const unsigned short* __restrict__ wb,
                 const float* __restrict__ bias,
                 float* __restrict__ f1sum) {
    extern __shared__ char smem[];
    unsigned short* xs = (unsigned short*)smem;
    float* f1part = (float*)(smem + XS_BYTES);

    const int tid = threadIdx.x;
    const int b = blockIdx.x / Hn;
    const int h = blockIdx.x % Hn;

    // Stage 3 input rows (h-1..h+1 incl. halo) into LDS: layout [w][row][ci pad 264].
    const uint4* xbq = (const uint4*)xb;
    uint4* xsq = (uint4*)xs;
    for (int it = tid; it < 3 * HP * 33; it += 256) {
        const int r = it / (HP * 33);
        const int rem = it - r * (HP * 33);
        const int w = rem / 33;
        const int q = rem - w * 33;
        uint4 v; v.x = v.y = v.z = v.w = 0u;
        if (q < 32) v = xbq[((size_t)(b * HP + h + r) * HP + w) * 32 + q];
        xsq[(w * 3 + r) * 33 + q] = v;
    }
    f1part[tid] = 0.f;
    __syncthreads();

    const int lane = tid & 31;
    const int wv   = tid >> 5;     // wave 0..7
    const int n    = lane & 15;    // N index (pixel) / M index (co) in frags
    const int hfh  = lane >> 4;    // lane half
    const int co0  = wv * 32;      // this wave owns co tiles [co0,co0+32)

    float bias0[8], bias1[8];
#pragma unroll
    for (int vi = 0; vi < 8; ++vi) {
        bias0[vi] = bias[co0 + hfh * 8 + vi];
        bias1[vi] = bias[co0 + 16 + hfh * 8 + vi];
    }

    const uint4* xq = (const uint4*)xs;

    // 4 pixel-tile accumulator pairs kept live: A fragments loaded ONCE per
    // (tap, K-chunk) and reused by all 4 pixel tiles (4x L2 traffic cut).
    v8f acc0[4], acc1[4];
#pragma unroll
    for (int pt = 0; pt < 4; ++pt) {
        acc0[pt] = v8f{0.f, 0.f, 0.f, 0.f, 0.f, 0.f, 0.f, 0.f};
        acc1[pt] = v8f{0.f, 0.f, 0.f, 0.f, 0.f, 0.f, 0.f, 0.f};
    }

    for (int s = 0; s < 9; ++s) {
        const int kh = s / 3;
        const int kw = s - 3 * kh;
        int eb[4];
#pragma unroll
        for (int pt = 0; pt < 4; ++pt) {
            int wc = pt * 16 + n + kw;
            if (wc > 57) wc = 57;                 // OOB pixels masked in epilogue
            eb[pt] = (wc * 3 + kh) * CIP;         // B-frag lds element base
        }
        const uint4* aq0 = (const uint4*)wb + ((size_t)(s * 256 + co0 + n) * 256) / 8;
        const uint4* aq1 = aq0 + (16 * 256) / 8;
        if (s < 8)                                 // hint next tap's weights
            __builtin_prefetch((const char*)((const uint4*)wb +
                ((size_t)((s + 1) * 256 + co0 + n) * 256) / 8), 0, 1);
#pragma unroll
        for (int cc = 0; cc < 256; cc += 32) {
            union { v16bf v; uint4 q[2]; } af0, af1;
            const int cia = cc + hfh * 8;          // A: K interleave 0-7/16-23
            af0.q[0] = aq0[cia >> 3];
            af0.q[1] = aq0[(cia + 16) >> 3];
            af1.q[0] = aq1[cia >> 3];
            af1.q[1] = aq1[(cia + 16) >> 3];
            const int cib = cc + hfh * 8;          // B: K interleave 0-7/16-23
#pragma unroll
            for (int pt = 0; pt < 4; ++pt) {
                union { v16bf v; uint4 q[2]; } bfr;
                bfr.q[0] = xq[(eb[pt] + cib) >> 3];
                bfr.q[1] = xq[(eb[pt] + cib + 16) >> 3];
                acc0[pt] = __builtin_amdgcn_wmma_f32_16x16x32_bf16(
                    false, af0.v, false, bfr.v, (short)0, acc0[pt], false, false);
                acc1[pt] = __builtin_amdgcn_wmma_f32_16x16x32_bf16(
                    false, af1.v, false, bfr.v, (short)0, acc1[pt], false, false);
            }
        }
    }

    // Epilogue: bias + relu + pixel-mask, reduce over the 16 N-lanes.
#pragma unroll
    for (int pt = 0; pt < 4; ++pt) {
        const bool valid = (pt * 16 + n) < Wn;
#pragma unroll
        for (int vi = 0; vi < 8; ++vi) {
            float r0 = valid ? fmaxf(acc0[pt][vi] + bias0[vi], 0.f) : 0.f;
            float r1 = valid ? fmaxf(acc1[pt][vi] + bias1[vi], 0.f) : 0.f;
#pragma unroll
            for (int m = 1; m < 16; m <<= 1) {
                r0 += __shfl_xor(r0, m, 16);
                r1 += __shfl_xor(r1, m, 16);
            }
            if (n == 0) {                              // lanes 0 and 16
                atomicAdd(&f1part[co0 + hfh * 8 + vi], r0);
                atomicAdd(&f1part[co0 + 16 + hfh * 8 + vi], r1);
            }
        }
    }
    __syncthreads();
    atomicAdd(&f1sum[b * Cn + tid], f1part[tid]);
}

// ---------------- tail: attention chain + CRF + head -----------------------
__global__ void k_tail(const float* __restrict__ f1sum,
                       const float* __restrict__ w0_1, const float* __restrict__ b0_1,
                       const float* __restrict__ w0_2, const float* __restrict__ b0_2,
                       const float* __restrict__ w0_3, const float* __restrict__ b0_3,
                       const float* __restrict__ w0_4, const float* __restrict__ b0_4,
                       const float* __restrict__ w1,   const float* __restrict__ b1,
                       const float* __restrict__ w2,   const float* __restrict__ b2,
                       const float* __restrict__ fc1,  const float* __restrict__ fc2,
                       const float* __restrict__ fc2b, const float* __restrict__ compat,
                       const float* __restrict__ sw,   float* __restrict__ out) {
    __shared__ float s0[256];   // f1c, later f4
    __shared__ float s1[256];   // f_c
    __shared__ float s2[256];   // f3
    __shared__ float s64a[64];  // f3s
    __shared__ float svs;
    __shared__ float sred[8];
    const int b = blockIdx.x, c = threadIdx.x;

    s0[c] = f1sum[b * Cn + c] * (1.f / (Hn * Wn));
    __syncthreads();

    float a2 = 0.f, avc = 0.f;                 // 1x1-spatial conv => center tap
    for (int ci = 0; ci < Cn; ++ci) {
        const float fv = s0[ci];
        a2  += w0_1[((size_t)c * Cn + ci) * 9 + 4] * fv;
        avc += fc1[(size_t)c * Cn + ci] * fv;
    }
    const float f2v = fmaxf(a2 + b0_1[c], 0.f);
    const float vc  = 1.f / (1.f + expf(-avc));
    s1[c] = vc * f2v;                          // f_c
    __syncthreads();

    float a3 = 0.f;
    for (int ci = 0; ci < Cn; ++ci) a3 += w0_2[((size_t)c * Cn + ci) * 9 + 4] * s1[ci];
    s2[c] = fmaxf(a3 + b0_2[c], 0.f);          // f3
    __syncthreads();

    float a4 = 0.f;
    for (int ci = 0; ci < Cn; ++ci) a4 += w0_3[((size_t)c * Cn + ci) * 9 + 4] * s2[ci];
    const float f4v = fmaxf(a4 + b0_3[c], 0.f);
    s0[c] = f4v;                               // f4 (s0 free after barrier 2)
    if (c < 64) {
        float as = 0.f;
        for (int ci = 0; ci < Cn; ++ci) as += w1[((size_t)c * Cn + ci) * 9 + 4] * s2[ci];
        s64a[c] = fmaxf(as + b1[c], 0.f);      // f3s
    }
    __syncthreads();

    if (c == 0) {                              // v0s + scalar CRF (1x1 spatial)
        float av = 0.f;
        for (int j = 0; j < 64; ++j) av += w2[j * 9 + 4] * s64a[j];
        const float v0s = fmaxf(av + b2[0], 0.f);
        const float u0 = v0s, u1 = -v0s;
        float m = fmaxf(u0, u1);
        float q0 = expf(u0 - m), q1 = expf(u1 - m);
        float zs = q0 + q1; q0 /= zs; q1 /= zs;
        const float c00 = compat[0], c01 = compat[1], c10 = compat[2], c11 = compat[3];
        const float sw0 = sw[0], sw1 = sw[1];
        for (int it = 0; it < 5; ++it) {
            const float msg0 = 0.25f * q0 * sw0;   // Gaussian center tap = 4/16
            const float msg1 = 0.25f * q1 * sw1;
            const float z0 = u0 - (c00 * msg0 + c01 * msg1);
            const float z1 = u1 - (c10 * msg0 + c11 * msg1);
            const float mm = fmaxf(z0, z1);
            q0 = expf(z0 - mm); q1 = expf(z1 - mm);
            const float s = q0 + q1; q0 /= s; q1 /= s;
        }
        svs = q1;                              // P(label=1)
    }
    __syncthreads();

    const float vsv = svs;
    float ar = 0.f;
    for (int ci = 0; ci < Cn; ++ci)
        ar += w0_4[((size_t)c * Cn + ci) * 9 + 4] * (vsv * s0[ci]);
    const float frv = fmaxf(ar + b0_4[c], 0.f);

    float val = fc2[c] * frv;
    for (int m = 1; m < 32; m <<= 1) val += __shfl_xor(val, m, 32);
    if ((c & 31) == 0) sred[c >> 5] = val;
    __syncthreads();
    if (c == 0) {
        float tot = 0.f;
        for (int i = 0; i < 8; ++i) tot += sred[i];
        out[b] = 1.f / (1.f + expf(-(tot + fc2b[0])));
    }
}

// ---------------------------------------------------------------------------
extern "C" void kernel_launch(void* const* d_in, const int* in_sizes, int n_in,
                              void* d_out, int out_size, void* d_ws, size_t ws_size,
                              hipStream_t stream) {
    (void)in_sizes; (void)n_in; (void)out_size; (void)ws_size;
    const float* x    = (const float*)d_in[0];
    const float* w0_0 = (const float*)d_in[1];
    const float* b0_0 = (const float*)d_in[2];
    const float* w0_1 = (const float*)d_in[3];
    const float* b0_1 = (const float*)d_in[4];
    const float* w0_2 = (const float*)d_in[5];
    const float* b0_2 = (const float*)d_in[6];
    const float* w0_3 = (const float*)d_in[7];
    const float* b0_3 = (const float*)d_in[8];
    const float* w0_4 = (const float*)d_in[9];
    const float* b0_4 = (const float*)d_in[10];
    const float* w1   = (const float*)d_in[11];
    const float* b1   = (const float*)d_in[12];
    const float* w2   = (const float*)d_in[13];
    const float* b2   = (const float*)d_in[14];
    const float* fc1  = (const float*)d_in[15];
    const float* fc2  = (const float*)d_in[16];
    const float* fc2b = (const float*)d_in[17];
    const float* cmp  = (const float*)d_in[18];
    const float* swv  = (const float*)d_in[19];

    unsigned short* wb   = (unsigned short*)d_ws;
    unsigned short* xbuf = (unsigned short*)((char*)d_ws + WB_BYTES);
    float*          f1s  = (float*)((char*)d_ws + WB_BYTES + XB_BYTES);

    // 1) zero halo ring of x-buffer + f1 partial sums
    k_zero_halo<<<Bn * 228, 32, 0, stream>>>(xbuf);
    k_zero<<<4, 256, 0, stream>>>((uint4*)f1s, (Bn * Cn * 4) / 16);
    // 2) repack inputs to bf16 layouts
    k_xprep<<<Bn * Hn * 4, 256, 0, stream>>>(x, xbuf);
    k_wprep<<<2304, 256, 0, stream>>>(w0_0, wb);
    // 3) dominant conv + relu + spatial pool (WMMA implicit GEMM)
    k_conv_pool<<<Bn * Hn, 256, SMEM_B, stream>>>(xbuf, wb, b0_0, f1s);
    // 4) tail chain per batch
    k_tail<<<Bn, 256, 0, stream>>>(f1s, w0_1, b0_1, w0_2, b0_2, w0_3, b0_3,
                                   w0_4, b0_4, w1, b1, w2, b2, fc1, fc2, fc2b,
                                   cmp, swv, (float*)d_out);
}
